// MambaBranch_1623497638604
// MI455X (gfx1250) — matrix operand order repaired and checked by reading, not compiled, not run on hardware
//
#include <hip/hip_runtime.h>
#include <hip/hip_bf16.h>
#include <stdint.h>

// ---------------- CDNA5 WMMA / TDM types ----------------
typedef __attribute__((ext_vector_type(16))) __bf16 v16bf;
typedef __attribute__((ext_vector_type(8)))  float  v8f;
typedef __attribute__((ext_vector_type(4)))  unsigned int v4u;
typedef __attribute__((ext_vector_type(8)))  int    v8i;
typedef __attribute__((ext_vector_type(4)))  int    v4i;

#define BATCH  4096
#define IN_DIM 480
#define DM     256
#define DI     512
#define NB     5
#define MT     32              // batch rows per workgroup
#define NWG    (BATCH / MT)    // 128 workgroups

// workspace layout (ushort/bf16 units)
#define OFF_PRE 0u
#define LEN_PRE (256u * 480u)
#define OFF_IN  (OFF_PRE + LEN_PRE)
#define LEN_IN  (5u * 1024u * 256u)
#define OFF_XP  (OFF_IN + LEN_IN)
#define LEN_XP  (5u * 48u * 512u)
#define OFF_DT  (OFF_XP + LEN_XP)
#define LEN_DT  (5u * 512u * 32u)       // K padded 16 -> 32
#define OFF_OUT (OFF_DT + LEN_DT)
#define LEN_OUT (5u * 256u * 512u)
#define OFF_CLS (OFF_OUT + LEN_OUT)
#define LEN_CLS (16u * 256u)            // N padded 8 -> 16

__device__ __forceinline__ unsigned short f2bf(float f) {
  unsigned u = __float_as_uint(f);
  u += 0x7FFFu + ((u >> 16) & 1u);      // round-to-nearest-even
  return (unsigned short)(u >> 16);
}
__device__ __forceinline__ float bf2f(unsigned short h) {
  return __uint_as_float(((unsigned)h) << 16);
}
__device__ __forceinline__ float siluf(float v) { return v / (1.0f + __expf(-v)); }
__device__ __forceinline__ float softplusf(float v) {
  return (v > 20.0f) ? v : log1pf(__expf(v));
}

struct Q32 { uint4 a, b; };  // 32 bytes = one v16bf fragment

// A fragment (16x32, M x K) from LDS, row-major with row stride `strideH` halves.
// Layout used here (matches frag_B): lane l -> row M = l&15; element e -> K = 16*(l>>4) + e (contiguous 32B).
__device__ __forceinline__ v16bf frag_A(const unsigned short* base, int strideH,
                                        int k0, int lane) {
  const int r = lane & 15, khi = lane >> 4;
  const unsigned short* p = base + r * strideH + k0 + khi * 16;
  Q32 t;
  t.a = *(const uint4*)(p);
  t.b = *(const uint4*)(p + 8);
  return __builtin_bit_cast(v16bf, t);
}

// B fragment (32x16, K x N) taken from row-major weight W[N][K] (B[k][n]=W[n][k]).
// Layout used here (matches frag_A): lane l -> col N = l&15; element e -> K = 16*(l>>4) + e (contiguous 32B).
__device__ __forceinline__ v16bf frag_B(const unsigned short* W, int K,
                                        int n0, int k0, int lane) {
  const int r = lane & 15, khi = lane >> 4;
  const unsigned short* p = W + (size_t)(n0 + r) * K + k0 + khi * 16;
  Q32 t;
  t.a = *(const uint4*)(p);
  t.b = *(const uint4*)(p + 8);
  return __builtin_bit_cast(v16bf, t);
}

__device__ __forceinline__ v8f wmma_bf(v16bf a, v16bf b, v8f c) {
  return __builtin_amdgcn_wmma_f32_16x16x32_bf16(false, a, false, b,
                                                 (short)0, c, false, false);
}

// Tensor Data Mover: copy n fp32 elements gsrc[i*stride] -> contiguous LDS.
// stride==1: 1D line copy; stride>1: 2D gather (tile_dim0=1, Y-stride=stride).
// Toolchain is the 6-arg builtin variant (clang-23 / therock headers).
__device__ __forceinline__ void tdm_1d(void* lds_dst, const float* gsrc,
                                       unsigned n, unsigned stride) {
  const unsigned lds_off = (unsigned)(size_t)lds_dst;   // low 32b = LDS offset
  const unsigned long long ga = (unsigned long long)(size_t)gsrc;
  unsigned t0, t1, tile0, tile1, s0;
  if (stride == 1) { t0 = n; t1 = 1; tile0 = n; tile1 = 1; s0 = n; }
  else             { t0 = 1; t1 = n; tile0 = 1; tile1 = n; s0 = stride; }
  v4u g0;
  g0.x = 1u;                                            // count=1, user mode
  g0.y = lds_off;                                       // D#.lds_addr (bytes)
  g0.z = (unsigned)ga;                                  // global_addr[31:0]
  g0.w = ((unsigned)(ga >> 32) & 0x01FFFFFFu) | (2u << 30);  // [56:32] | type=2
  v8i g1;
  g1[0] = (int)(2u << 16);                              // data_size = 4B
  g1[1] = (int)((t0 & 0xFFFFu) << 16);                  // tensor_dim0[15:0]
  g1[2] = (int)((t0 >> 16) | ((t1 & 0xFFFFu) << 16));   // dim0 hi | dim1 lo
  g1[3] = (int)((t1 >> 16) | (tile0 << 16));            // dim1 hi | tile_dim0
  g1[4] = (int)tile1;                                   // tile_dim1 (tile_dim2=0)
  g1[5] = (int)s0;                                      // tensor_dim0_stride
  g1[6] = 0;
  g1[7] = 0;
  const v4i z4 = {0, 0, 0, 0};
  const v8i z8 = {0, 0, 0, 0, 0, 0, 0, 0};
  __builtin_amdgcn_tensor_load_to_lds(g0, g1, z4, z4, z8, 0);
}

// ---------------- weight prep kernels (fp32 -> bf16 in workspace) ----------------
__global__ void k_cvt(const float* __restrict__ s, unsigned short* __restrict__ d, int n) {
  int i = blockIdx.x * blockDim.x + threadIdx.x;
  if (i < n) d[i] = f2bf(s[i]);
}
// dt_w (5,512,16) -> (5,512,32), K 16..31 zero
__global__ void k_cvt_dtw(const float* __restrict__ s, unsigned short* __restrict__ d) {
  int i = blockIdx.x * blockDim.x + threadIdx.x;
  if (i >= 5 * 512 * 32) return;
  int k = i & 31;
  int row = i >> 5;
  d[i] = (k < 16) ? f2bf(s[row * 16 + k]) : (unsigned short)0;
}
// cls_w (8,256) -> (16,256), rows 8..15 zero
__global__ void k_cvt_cls(const float* __restrict__ s, unsigned short* __restrict__ d) {
  int i = blockIdx.x * blockDim.x + threadIdx.x;
  if (i >= 16 * 256) return;
  int n = i >> 8;
  d[i] = (n < 8) ? f2bf(s[i]) : (unsigned short)0;
}

// ---------------- fused persistent-tile Mamba kernel ----------------
__global__ __launch_bounds__(256, 1)
void k_mamba_fused(const float* __restrict__ x,
                   const float* __restrict__ pre_b,
                   const float* __restrict__ conv_w,   // (5,512,1,4)
                   const float* __restrict__ conv_b,   // (5,512)
                   const float* __restrict__ dt_b,     // (5,512)
                   const float* __restrict__ Dvec,     // (5,512)
                   const float* __restrict__ norm_g,
                   const float* __restrict__ norm_b,
                   const float* __restrict__ cls_b,
                   const unsigned short* __restrict__ ws,
                   float* __restrict__ out) {
  __shared__ __align__(16) float          sh_h[MT * DM];        // fp32 residual
  __shared__ __align__(16) unsigned short sh_a[MT * DI];        // bf16 A staging
  __shared__ __align__(16) unsigned short sh_xs[MT * DI];       // bf16 xs
  __shared__ __align__(16) unsigned short sh_z[MT * DI];        // bf16 z
  __shared__ __align__(16) unsigned short sh_dtp[MT * 32];      // bf16 dtp (K-padded)
  __shared__ float sh_bm[MT * 16];
  __shared__ float sh_cm[MT * 16];
  __shared__ float sh_s[MT];
  // per-block parameter vectors staged by the Tensor Data Mover
  __shared__ __align__(16) float sh_cw3[DI];
  __shared__ __align__(16) float sh_cb[DI];
  __shared__ __align__(16) float sh_dtb[DI];
  __shared__ __align__(16) float sh_D[DI];

  const int tid  = threadIdx.x;
  const int lane = tid & 31;
  const int wave = tid >> 5;
  const int b0   = blockIdx.x * MT;

  const unsigned short* Wpre = ws + OFF_PRE;
  const unsigned short* Win  = ws + OFF_IN;
  const unsigned short* Wxp  = ws + OFF_XP;
  const unsigned short* Wdt  = ws + OFF_DT;
  const unsigned short* Wout = ws + OFF_OUT;
  const unsigned short* Wcls = ws + OFF_CLS;

  // zero dtp pad (cols 16..31 stay zero for the whole kernel)
  for (int i = tid; i < MT * 32; i += 256) sh_dtp[i] = 0;

  // load x tile (32 x 480 fp32) -> bf16 in sh_a, row stride IN_DIM
  {
    const float4* xg = (const float4*)(x + (size_t)b0 * IN_DIM);
    for (int i = tid; i < MT * IN_DIM / 4; i += 256) {
      float4 v = xg[i];
      int o = i * 4;
      sh_a[o + 0] = f2bf(v.x); sh_a[o + 1] = f2bf(v.y);
      sh_a[o + 2] = f2bf(v.z); sh_a[o + 3] = f2bf(v.w);
    }
  }
  __syncthreads();

  const int mt    = wave & 1;       // m-tile (16 rows) 0/1
  const int ng    = wave >> 1;      // n-group 0..3
  const int col   = lane & 15;
  const int rbase = (lane >> 4) * 8;

  // ---- pre: h = x @ pre_w^T + pre_b   (N=256, K=480) ----
  {
    v8f acc[4];
#pragma unroll
    for (int nt = 0; nt < 4; ++nt) acc[nt] = (v8f){0.f,0.f,0.f,0.f,0.f,0.f,0.f,0.f};
    for (int kc = 0; kc < 3; ++kc) {           // 3 chunks x 5 k-steps of 32
      v16bf af[5];
#pragma unroll
      for (int kk = 0; kk < 5; ++kk)
        af[kk] = frag_A(sh_a + mt * 16 * IN_DIM, IN_DIM, kc * 160 + kk * 32, lane);
#pragma unroll
      for (int nt = 0; nt < 4; ++nt) {
        const int n0 = ng * 64 + nt * 16;
#pragma unroll
        for (int kk = 0; kk < 5; ++kk)
          acc[nt] = wmma_bf(af[kk], frag_B(Wpre, IN_DIM, n0, kc * 160 + kk * 32, lane),
                            acc[nt]);
      }
    }
#pragma unroll
    for (int nt = 0; nt < 4; ++nt) {
      const int n0 = ng * 64 + nt * 16;
#pragma unroll
      for (int r = 0; r < 8; ++r) {
        const int row = mt * 16 + rbase + r;
        const int n = n0 + col;
        sh_h[row * DM + n] = acc[nt][r] + pre_b[n];
      }
    }
  }
  __syncthreads();

  for (int blk = 0; blk < NB; ++blk) {
    // kick off TDM staging of this block's parameter vectors (overlaps B0)
    if (wave == 0) {
      tdm_1d(sh_cw3, conv_w + (size_t)blk * DI * 4 + 3, DI, 4);  // strided gather
      tdm_1d(sh_cb,  conv_b + (size_t)blk * DI, DI, 1);
      tdm_1d(sh_dtb, dt_b   + (size_t)blk * DI, DI, 1);
      tdm_1d(sh_D,   Dvec   + (size_t)blk * DI, DI, 1);
    }

    // B0: h -> bf16 A (stride DM)
    for (int i = tid; i < MT * DM; i += 256) sh_a[i] = f2bf(sh_h[i]);
    if (wave == 0) __builtin_amdgcn_s_wait_tensorcnt(0);
    __syncthreads();

    // B1: in_proj (N=1024, K=256) fused with conv-tap + SiLU / z split.
    // A fragments hoisted to registers, reused across 16 n-tiles.
    const unsigned short* W1 = Win + (size_t)blk * 1024u * 256u;
    {
      v16bf af[8];
#pragma unroll
      for (int kk = 0; kk < 8; ++kk)
        af[kk] = frag_A(sh_a + mt * 16 * DM, DM, kk * 32, lane);
      for (int nt = 0; nt < 16; ++nt) {
        const int n0 = ng * 256 + nt * 16;
        v8f acc = {0.f,0.f,0.f,0.f,0.f,0.f,0.f,0.f};
#pragma unroll
        for (int kk = 0; kk < 8; ++kk)
          acc = wmma_bf(af[kk], frag_B(W1, DM, n0, kk * 32, lane), acc);
#pragma unroll
        for (int r = 0; r < 8; ++r) {
          const int row = mt * 16 + rbase + r;
          const int n = n0 + col;
          float v = acc[r];
          if (n < DI) {                    // xs: last conv tap + bias + SiLU
            v = siluf(v * sh_cw3[n] + sh_cb[n]);
            sh_xs[row * DI + n] = f2bf(v);
          } else {                         // z
            sh_z[row * DI + (n - DI)] = f2bf(v);
          }
        }
      }
    }
    __syncthreads();

    // B2: x_proj (N=48, K=512) -> dtp / Bm / Cm  (waves 0..5: 2 mtiles x 3 ntiles)
    const unsigned short* W2 = Wxp + (size_t)blk * 48u * 512u;
    if (wave < 6) {
      const int mt2 = wave & 1;
      const int n0  = (wave >> 1) * 16;
      v8f acc = {0.f,0.f,0.f,0.f,0.f,0.f,0.f,0.f};
      for (int k0 = 0; k0 < DI; k0 += 32) {
        v16bf a = frag_A(sh_xs + mt2 * 16 * DI, DI, k0, lane);
        acc = wmma_bf(a, frag_B(W2, DI, n0, k0, lane), acc);
      }
#pragma unroll
      for (int r = 0; r < 8; ++r) {
        const int row = mt2 * 16 + rbase + r;
        const int f = n0 + col;
        const float v = acc[r];
        if (f < 16)       sh_dtp[row * 32 + f]       = f2bf(v);
        else if (f < 32)  sh_bm[row * 16 + (f - 16)] = v;
        else              sh_cm[row * 16 + (f - 32)] = v;
      }
    }
    __syncthreads();

    // s[b] = Bm . Cm  (single-step SSM collapses the scan to this dot product)
    if (tid < MT) {
      float s = 0.f;
#pragma unroll
      for (int n = 0; n < 16; ++n) s += sh_bm[tid * 16 + n] * sh_cm[tid * 16 + n];
      sh_s[tid] = s;
    }
    __syncthreads();

    // B3: dt GEMM (N=512, K=32 padded, single WMMA per tile) fused with
    //     softplus and y = (dt*s + D) * xs * silu(z); y -> bf16 in sh_a.
    const unsigned short* W3 = Wdt + (size_t)blk * 512u * 32u;
    {
      const v16bf adt = frag_A(sh_dtp + mt * 16 * 32, 32, 0, lane);
      for (int nt = 0; nt < 8; ++nt) {
        const int n0 = ng * 128 + nt * 16;
        v8f acc = {0.f,0.f,0.f,0.f,0.f,0.f,0.f,0.f};
        acc = wmma_bf(adt, frag_B(W3, 32, n0, 0, lane), acc);
#pragma unroll
        for (int r = 0; r < 8; ++r) {
          const int row = mt * 16 + rbase + r;
          const int n = n0 + col;
          const float dt = softplusf(acc[r] + sh_dtb[n]);
          const float xs = bf2f(sh_xs[row * DI + n]);
          const float z  = bf2f(sh_z[row * DI + n]);
          const float y  = (dt * sh_s[row] + sh_D[n]) * xs * siluf(z);
          sh_a[row * DI + n] = f2bf(y);
        }
      }
    }
    __syncthreads();

    // B4: out_proj (N=256, K=512) + residual into sh_h.
    // K split into 2 register-resident chunks; 4 accumulators live.
    const unsigned short* W4 = Wout + (size_t)blk * 256u * 512u;
    {
      v8f acc[4];
#pragma unroll
      for (int nt = 0; nt < 4; ++nt) acc[nt] = (v8f){0.f,0.f,0.f,0.f,0.f,0.f,0.f,0.f};
      for (int kc = 0; kc < 2; ++kc) {
        v16bf af[8];
#pragma unroll
        for (int kk = 0; kk < 8; ++kk)
          af[kk] = frag_A(sh_a + mt * 16 * DI, DI, kc * 256 + kk * 32, lane);
#pragma unroll
        for (int nt = 0; nt < 4; ++nt) {
          const int n0 = ng * 64 + nt * 16;
#pragma unroll
          for (int kk = 0; kk < 8; ++kk)
            acc[nt] = wmma_bf(af[kk], frag_B(W4, DI, n0, kc * 256 + kk * 32, lane),
                              acc[nt]);
        }
      }
#pragma unroll
      for (int nt = 0; nt < 4; ++nt) {
        const int n0 = ng * 64 + nt * 16;
#pragma unroll
        for (int r = 0; r < 8; ++r) {
          const int row = mt * 16 + rbase + r;
          const int n = n0 + col;
          sh_h[row * DM + n] += acc[nt][r];   // unique owner: no race
        }
      }
    }
    __syncthreads();
  }

  // LayerNorm over DM, write bf16 into sh_a (stride DM)
  if (tid < MT) {
    const float* hr = sh_h + tid * DM;
    float mu = 0.f;
    for (int k = 0; k < DM; ++k) mu += hr[k];
    mu *= (1.0f / DM);
    float var = 0.f;
    for (int k = 0; k < DM; ++k) { const float d = hr[k] - mu; var += d * d; }
    var *= (1.0f / DM);
    const float rs = rsqrtf(var + 1e-5f);
    for (int k = 0; k < DM; ++k)
      sh_a[tid * DM + k] = f2bf((hr[k] - mu) * rs * norm_g[k] + norm_b[k]);
  }
  __syncthreads();

  // classifier: hn @ cls_w^T + cls_b  (N padded to 16, K=256); waves 0,1
  if (wave < 2) {
    v8f acc = {0.f,0.f,0.f,0.f,0.f,0.f,0.f,0.f};
    for (int k0 = 0; k0 < DM; k0 += 32) {
      v16bf a = frag_A(sh_a + wave * 16 * DM, DM, k0, lane);
      acc = wmma_bf(a, frag_B(Wcls, DM, 0, k0, lane), acc);
    }
#pragma unroll
    for (int r = 0; r < 8; ++r) {
      const int row = wave * 16 + rbase + r;
      if (col < 8)
        out[(size_t)(b0 + row) * 8 + col] = acc[r] + cls_b[col];
    }
  }
}

// ---------------- launch ----------------
extern "C" void kernel_launch(void* const* d_in, const int* in_sizes, int n_in,
                              void* d_out, int out_size, void* d_ws, size_t ws_size,
                              hipStream_t stream) {
  const float* x      = (const float*)d_in[0];
  const float* pre_w  = (const float*)d_in[1];
  const float* pre_b  = (const float*)d_in[2];
  const float* in_w   = (const float*)d_in[3];
  const float* conv_w = (const float*)d_in[4];
  const float* conv_b = (const float*)d_in[5];
  const float* xp_w   = (const float*)d_in[6];
  const float* dt_w   = (const float*)d_in[7];
  const float* dt_b   = (const float*)d_in[8];
  // d_in[9] = A_log: provably dead for L=1 (state starts at zero)
  const float* Dv     = (const float*)d_in[10];
  const float* out_w  = (const float*)d_in[11];
  const float* ng     = (const float*)d_in[12];
  const float* nb     = (const float*)d_in[13];
  const float* cls_w  = (const float*)d_in[14];
  const float* cls_b  = (const float*)d_in[15];
  float* out          = (float*)d_out;
  unsigned short* ws  = (unsigned short*)d_ws;

  auto blocks = [](unsigned n) { return (n + 255u) / 256u; };

  k_cvt<<<blocks(LEN_PRE), 256, 0, stream>>>(pre_w, ws + OFF_PRE, (int)LEN_PRE);
  k_cvt<<<blocks(LEN_IN),  256, 0, stream>>>(in_w,  ws + OFF_IN,  (int)LEN_IN);
  k_cvt<<<blocks(LEN_XP),  256, 0, stream>>>(xp_w,  ws + OFF_XP,  (int)LEN_XP);
  k_cvt<<<blocks(LEN_OUT), 256, 0, stream>>>(out_w, ws + OFF_OUT, (int)LEN_OUT);
  k_cvt_dtw<<<blocks(LEN_DT), 256, 0, stream>>>(dt_w, ws + OFF_DT);
  k_cvt_cls<<<blocks(LEN_CLS), 256, 0, stream>>>(cls_w, ws + OFF_CLS);

  k_mamba_fused<<<NWG, 256, 0, stream>>>(x, pre_b, conv_w, conv_b, dt_b, Dv,
                                         ng, nb, cls_b, ws, out);
}
